// OrientedRPN_84464826843250
// MI455X (gfx1250) — matrix operand order repaired and checked by reading, not compiled, not run on hardware
//
#include <hip/hip_runtime.h>
#include <hip/hip_bf16.h>
#include <cstddef>
#include <cstdint>

// ---------------------------------------------------------------------------
// Types for CDNA5 WMMA (wave32): v_wmma_f32_16x16x32_bf16
// ---------------------------------------------------------------------------
typedef __attribute__((ext_vector_type(16))) __bf16 v16bf;
typedef __attribute__((ext_vector_type(8)))  __bf16 v8bf;
typedef __attribute__((ext_vector_type(8)))  float  v8f;

union Frag16 { v16bf v; v8bf h[2]; };

__device__ __forceinline__ void split_f32(float x, __bf16& hi, __bf16& lo) {
  __bf16 h = (__bf16)x;
  hi = h;
  lo = (__bf16)(x - (float)h);
}

// Async DMA: global -> LDS, 16B per lane, tracked with ASYNCcnt (CDNA5).
__device__ __forceinline__ void async_copy_b128(unsigned lds_off, const void* gsrc) {
  asm volatile("global_load_async_to_lds_b128 %0, %1, off"
               :: "v"(lds_off), "v"(gsrc) : "memory");
}
__device__ __forceinline__ unsigned lds_addr_of(const void* p) {
  return (unsigned)(uintptr_t)p;   // generic->LDS aperture: low 32 bits = offset
}

// ---------------------------------------------------------------------------
// Kernel 0: pre-tile conv weights into WMMA-friendly bf16 hi/lo layout.
// pw[kk][plane][oc][kl] halves, kk = tap*8 + icg (72 chunks of K=32).
// Each 64-oc x 32-k tile is a contiguous 4KB block -> async-copyable.
// grid = 72, block = 256 (one oc row each).
// ---------------------------------------------------------------------------
__global__ __launch_bounds__(256) void prep_weights(
    const float* __restrict__ wc, void* __restrict__ pwv)
{
  __bf16* pw = (__bf16*)pwv;
  const int kk  = blockIdx.x;          // 0..71
  const int tap = kk >> 3;
  const int c0  = (kk & 7) << 5;
  const int oc  = threadIdx.x;

  const float* src = wc + (size_t)oc * 2304 + (size_t)c0 * 9 + tap;
  __bf16* dh = pw + ((size_t)(kk * 2) * 256 + oc) * 32;        // hi plane
  __bf16* dl = dh + (size_t)256 * 32;                          // lo plane
  #pragma unroll
  for (int i = 0; i < 32; ++i) {
    float v = src[(size_t)i * 9];
    __bf16 hi, lo; split_f32(v, hi, lo);
    dh[i] = hi; dl[i] = lo;
  }
}

// ---------------------------------------------------------------------------
// Kernel 1: 3x3 conv (256->256, pad SAME) as implicit GEMM.
// M = 2*H*W (row = (b,oh,ow)), N = 256 (oc), K = 9*256 (tap-major, ic within).
// Block: 256 threads (8 waves), tile 64(M) x 64(N). Wave w: M-sub = w&3,
// N-subs = (w>>2)*2 + {0,1}. Split-precision bf16 (hi/lo), fp32 accumulate.
// B tiles double-buffered in LDS via GLOBAL_LOAD_ASYNC_TO_LDS_B128.
// ---------------------------------------------------------------------------
__global__ __launch_bounds__(256) void conv3x3_wmma(
    const float* __restrict__ x, const void* __restrict__ pwv,
    const float* __restrict__ bc, float* __restrict__ xout,
    int H, int W)
{
  const int hw = H * W;
  const int tile_n = blockIdx.x & 3;
  const int tile_m = blockIdx.x >> 2;
  const int tid  = threadIdx.x;
  const int lane = tid & 31;
  const int wave = tid >> 5;
  const int wm   = wave & 3;
  const int wg   = wave >> 2;
  const int lh   = lane & 15;
  const int lhi  = lane >> 4;   // 0 or 1 (half-wave)

  __shared__ __align__(16) __bf16 Ah[64][40];
  __shared__ __align__(16) __bf16 Al[64][40];
  __shared__ __align__(16) __bf16 Bbuf[2][2][64][32];   // [buf][plane][oc][k]

  v8f acc0 = {};
  v8f acc1 = {};

  // A-tile loader mapping: consecutive tid -> consecutive spatial m (coalesced)
  const int am = tid & 63;
  const int at = tid >> 6;                 // 4 groups of 8 ic each
  const int mg = tile_m * 64 + am;
  const int bb = mg / hw;
  const int pp = mg - bb * hw;
  const int oh = pp / W;
  const int ow = pp - oh * W;

  const char* pw = (const char*)pwv;
  // Issue async DMA of B tile for chunk kkx into Bbuf[buf]: 2 x B128 per lane.
  auto issueB = [&](int kkx, int buf) {
    const size_t base = (size_t)kkx * 32768 + (size_t)tile_n * 4096 + (size_t)tid * 16;
    async_copy_b128(lds_addr_of(&Bbuf[buf][0][0][0]) + (unsigned)(tid * 16), pw + base);
    async_copy_b128(lds_addr_of(&Bbuf[buf][1][0][0]) + (unsigned)(tid * 16), pw + base + 16384);
  };

  issueB(0, 0);   // prologue

  for (int kk = 0; kk < 72; ++kk) {        // K = 72 chunks of 32
    const int cur = kk & 1;
    const int tap = kk >> 3;
    const int c0  = (kk & 7) << 5;
    const int dh = tap / 3 - 1;
    const int dw = tap % 3 - 1;
    const int ih = oh + dh;
    const int iw = ow + dw;
    const bool valid = ((unsigned)ih < (unsigned)H) && ((unsigned)iw < (unsigned)W);

    __syncthreads();   // all fragment reads of previous iteration completed
    {
      const float* src = x + (((size_t)bb * 256 + (size_t)(c0 + at * 8)) * H + ih) * W + iw;
      #pragma unroll
      for (int i = 0; i < 8; ++i) {
        float v = valid ? src[(size_t)i * hw] : 0.0f;
        __bf16 hi, lo; split_f32(v, hi, lo);
        Ah[am][at * 8 + i] = hi;
        Al[am][at * 8 + i] = lo;
      }
    }
    // Prefetch next B tile into the back buffer, then wait for current tile
    // (<=2 leaves only the just-issued pair in flight).
    if (kk + 1 < 72) {
      issueB(kk + 1, cur ^ 1);
      asm volatile("s_wait_asynccnt 2" ::: "memory");
    } else {
      asm volatile("s_wait_asynccnt 0" ::: "memory");
    }
    __syncthreads();   // A stores visible; all waves' B DMA slices landed

    // A fragment (ISA 16-bit A 16x32 layout): lane l, elem i ->
    //   M = l%16, K = (i<8?0:16) + (l>=16?8:0) + i%8
    const int amrow = wm * 16 + lh;
    const int ak0   = lhi * 8;
    Frag16 a_hi, a_lo;
    a_hi.h[0] = *(const v8bf*)&Ah[amrow][ak0];
    a_hi.h[1] = *(const v8bf*)&Ah[amrow][16 + ak0];
    a_lo.h[0] = *(const v8bf*)&Al[amrow][ak0];
    a_lo.h[1] = *(const v8bf*)&Al[amrow][16 + ak0];

    #pragma unroll
    for (int j = 0; j < 2; ++j) {
      // B fragment (32x16): lane l, elem i -> K = (i<8?0:16) + (l>=16?8:0) + i%8, N = l%16
      const int nrow = (wg * 2 + j) * 16 + lh;
      const int bk0  = lhi * 8;
      Frag16 b_hi, b_lo;
      b_hi.h[0] = *(const v8bf*)&Bbuf[cur][0][nrow][bk0];
      b_hi.h[1] = *(const v8bf*)&Bbuf[cur][0][nrow][bk0 + 16];
      b_lo.h[0] = *(const v8bf*)&Bbuf[cur][1][nrow][bk0];
      b_lo.h[1] = *(const v8bf*)&Bbuf[cur][1][nrow][bk0 + 16];
      v8f& acc = j ? acc1 : acc0;
      acc = __builtin_amdgcn_wmma_f32_16x16x32_bf16(false, a_hi.v, false, b_hi.v,
                                                    (short)0, acc, false, false);
      acc = __builtin_amdgcn_wmma_f32_16x16x32_bf16(false, a_hi.v, false, b_lo.v,
                                                    (short)0, acc, false, false);
      acc = __builtin_amdgcn_wmma_f32_16x16x32_bf16(false, a_lo.v, false, b_hi.v,
                                                    (short)0, acc, false, false);
    }
  }

  // C/D layout: VGPR r, lanes 0-15: M=r, N=lane; lanes 16-31: M=8+r, N=lane-16
  #pragma unroll
  for (int j = 0; j < 2; ++j) {
    const v8f& acc = j ? acc1 : acc0;
    const int oc = tile_n * 64 + (wg * 2 + j) * 16 + lh;
    const float bias = bc[oc];
    #pragma unroll
    for (int r = 0; r < 8; ++r) {
      const int m = tile_m * 64 + wm * 16 + (lhi ? 8 : 0) + r;
      xout[(size_t)m * 256 + oc] = acc[r] + bias;
    }
  }
}

// ---------------------------------------------------------------------------
// Kernel 2: fused 1x1 heads. Y[M][32] = X[M][256] @ Wh^T, rows 0-17 = reg_w,
// rows 18-20 = obj_w, rows 21-31 = 0. Scatter into reference layouts.
// Block: 128 threads (4 waves), tile 64(M) x 32(N).
// ---------------------------------------------------------------------------
__global__ __launch_bounds__(128) void heads_wmma(
    const float* __restrict__ xin, const float* __restrict__ regw,
    const float* __restrict__ regb, const float* __restrict__ objw,
    const float* __restrict__ objb, float* __restrict__ regout,
    float* __restrict__ objout, int hw)
{
  const int tile_m = blockIdx.x;
  const int tid  = threadIdx.x;
  const int lane = tid & 31;
  const int wm   = tid >> 5;     // wave 0..3 -> M sub-tile
  const int lh   = lane & 15;
  const int lhi  = lane >> 4;

  __shared__ __align__(16) __bf16 Ah[64][40];
  __shared__ __align__(16) __bf16 Al[64][40];
  __shared__ __align__(16) __bf16 Bh[32][40];
  __shared__ __align__(16) __bf16 Bl[32][40];

  v8f acc0 = {};
  v8f acc1 = {};

  const int kl = tid & 31;
  const int tg = tid >> 5;       // 0..3

  for (int kk = 0; kk < 8; ++kk) {
    const int c0 = kk << 5;
    __syncthreads();
    // A: 64x32, coalesced over k
    #pragma unroll
    for (int q = 0; q < 16; ++q) {
      const int ml = tg * 16 + q;
      float v = xin[(size_t)(tile_m * 64 + ml) * 256 + c0 + kl];
      __bf16 hi, lo; split_f32(v, hi, lo);
      Ah[ml][kl] = hi; Al[ml][kl] = lo;
    }
    // B: 32x32
    #pragma unroll
    for (int i = 0; i < 8; ++i) {
      const int n = tg * 8 + i;
      float v = 0.0f;
      if (n < 18)      v = regw[(size_t)n * 256 + c0 + kl];
      else if (n < 21) v = objw[(size_t)(n - 18) * 256 + c0 + kl];
      __bf16 hi, lo; split_f32(v, hi, lo);
      Bh[n][kl] = hi; Bl[n][kl] = lo;
    }
    __syncthreads();

    const int amrow = wm * 16 + lh;
    const int ak0   = lhi * 8;
    Frag16 a_hi, a_lo;
    a_hi.h[0] = *(const v8bf*)&Ah[amrow][ak0];
    a_hi.h[1] = *(const v8bf*)&Ah[amrow][16 + ak0];
    a_lo.h[0] = *(const v8bf*)&Al[amrow][ak0];
    a_lo.h[1] = *(const v8bf*)&Al[amrow][16 + ak0];

    #pragma unroll
    for (int j = 0; j < 2; ++j) {
      const int nrow = j * 16 + lh;
      const int bk0  = lhi * 8;
      Frag16 b_hi, b_lo;
      b_hi.h[0] = *(const v8bf*)&Bh[nrow][bk0];
      b_hi.h[1] = *(const v8bf*)&Bh[nrow][bk0 + 16];
      b_lo.h[0] = *(const v8bf*)&Bl[nrow][bk0];
      b_lo.h[1] = *(const v8bf*)&Bl[nrow][bk0 + 16];
      v8f& acc = j ? acc1 : acc0;
      acc = __builtin_amdgcn_wmma_f32_16x16x32_bf16(false, a_hi.v, false, b_hi.v,
                                                    (short)0, acc, false, false);
      acc = __builtin_amdgcn_wmma_f32_16x16x32_bf16(false, a_hi.v, false, b_lo.v,
                                                    (short)0, acc, false, false);
      acc = __builtin_amdgcn_wmma_f32_16x16x32_bf16(false, a_lo.v, false, b_hi.v,
                                                    (short)0, acc, false, false);
    }
  }

  #pragma unroll
  for (int j = 0; j < 2; ++j) {
    const v8f& acc = j ? acc1 : acc0;
    const int n = j * 16 + lh;
    if (n < 21) {
      const float bias = (n < 18) ? regb[n] : objb[n - 18];
      #pragma unroll
      for (int r = 0; r < 8; ++r) {
        const int m = tile_m * 64 + wm * 16 + (lhi ? 8 : 0) + r;
        const int b = m / hw;
        const int p = m - b * hw;
        const float val = acc[r] + bias;
        if (n < 18) {
          const int a = n / 6, c = n - a * 6;
          regout[((size_t)(b * 3 + a) * hw + p) * 6 + c] = val;
        } else {
          objout[(size_t)(b * 3 + (n - 18)) * hw + p] = val;
        }
      }
    }
  }
}

// ---------------------------------------------------------------------------
// Kernel 3: per (b, channel) mean / population-std over na*h*w offsets.
// grid = 12 (b*6), block = 256.
// ---------------------------------------------------------------------------
__global__ __launch_bounds__(256) void stats_kernel(
    const float* __restrict__ reg, float* __restrict__ stats, int hw)
{
  const int bb = blockIdx.x / 6;
  const int c  = blockIdx.x - bb * 6;
  const int N3 = 3 * hw;
  const int tid = threadIdx.x;

  __shared__ float sS[256];
  __shared__ float sQ[256];

  float s = 0.0f, q = 0.0f;
  for (int j = tid; j < N3; j += 256) {
    float v = reg[((size_t)bb * N3 + j) * 6 + c];
    s += v; q += v * v;
  }
  sS[tid] = s; sQ[tid] = q;
  __syncthreads();
  for (int off = 128; off > 0; off >>= 1) {
    if (tid < off) { sS[tid] += sS[tid + off]; sQ[tid] += sQ[tid + off]; }
    __syncthreads();
  }
  if (tid == 0) {
    const float inv = 1.0f / (float)N3;
    const float mu  = sS[0] * inv;
    float var = sQ[0] * inv - mu * mu;
    var = var > 0.0f ? var : 0.0f;
    stats[(bb * 6 + c) * 2 + 0] = mu;
    stats[(bb * 6 + c) * 2 + 1] = sqrtf(var);
  }
}

// ---------------------------------------------------------------------------
// Kernel 4: exact top-K with stable tie-break (lowest index first, matching
// lax.top_k). Destroys `sc`. grid = 2 (batch), block = 1024.
// ---------------------------------------------------------------------------
__global__ __launch_bounds__(1024) void topk_kernel(
    float* __restrict__ sc, int* __restrict__ oidx, float* __restrict__ osc,
    int N, int K)
{
  const int bb = blockIdx.x;
  sc   += (size_t)bb * N;
  oidx += (size_t)bb * K;
  osc  += (size_t)bb * K;

  __shared__ float lv[1024];
  __shared__ int   li[1024];

  const int tid   = threadIdx.x;
  const int chunk = (N + 1023) >> 10;
  const int s0 = tid * chunk;
  int s1 = s0 + chunk; if (s1 > N) s1 = N;

  const float NEG = -__builtin_inff();
  float bv = NEG; int bi = 0x7fffffff;
  for (int j = s0; j < s1; ++j) {
    float v = sc[j];
    if (v > bv) { bv = v; bi = j; }   // strict > keeps lowest index on ties
  }

  for (int it = 0; it < K; ++it) {
    lv[tid] = bv; li[tid] = bi;
    __syncthreads();
    for (int off = 512; off > 0; off >>= 1) {
      if (tid < off) {
        float v2 = lv[tid + off]; int i2 = li[tid + off];
        if (v2 > lv[tid] || (v2 == lv[tid] && i2 < li[tid])) {
          lv[tid] = v2; li[tid] = i2;
        }
      }
      __syncthreads();
    }
    const float wv = lv[0];
    const int   wi = li[0];
    if (tid == 0) { oidx[it] = wi; osc[it] = wv; }
    if (wi >= s0 && wi < s1) {   // owner removes winner and rescans its chunk
      sc[wi] = NEG;
      bv = NEG; bi = 0x7fffffff;
      for (int j = s0; j < s1; ++j) {
        float v = sc[j];
        if (v > bv) { bv = v; bi = j; }
      }
    }
    __syncthreads();
  }
}

// ---------------------------------------------------------------------------
// Kernel 5: normalize + decode vertices + HBB for the K selected anchors,
// sequential-greedy NMS (exact reference recurrence), write final output.
// grid = 2 (batch), block = 256. K <= 2000.
// ---------------------------------------------------------------------------
__global__ __launch_bounds__(256) void decode_nms_kernel(
    const float* __restrict__ reg, const float* __restrict__ stats,
    const int* __restrict__ kidx, const float* __restrict__ kscore,
    float* __restrict__ verts_ws, float* __restrict__ out,
    int hw, int W_, int K, int out_off)
{
  const int bb  = blockIdx.x;
  const int tid = threadIdx.x;
  reg      += (size_t)bb * 3 * hw * 6;
  kidx     += (size_t)bb * K;
  kscore   += (size_t)bb * K;
  verts_ws += (size_t)bb * K * 8;

  __shared__ float sHbb[2000][4];
  __shared__ int   sKeep[2000];

  const float AW[3]  = {5.65685424949238f, 8.0f, 11.31370849898476f};
  const float AH[3]  = {11.31370849898476f, 8.0f, 5.65685424949238f};
  const float TGT[6] = {1.0f, 1.0f, 1.0f, 1.0f, 0.5f, 0.5f};

  float mu[6], sd[6];
  #pragma unroll
  for (int c = 0; c < 6; ++c) {
    mu[c] = stats[(bb * 6 + c) * 2 + 0];
    sd[c] = stats[(bb * 6 + c) * 2 + 1];
  }

  // Phase A: decode selected anchors
  for (int k = tid; k < K; k += 256) {
    const int i  = kidx[k];
    const int a  = i / hw;
    const int p  = i - a * hw;
    const int oh = p / W_;
    const int ow = p - oh * W_;

    float o[6];
    #pragma unroll
    for (int c = 0; c < 6; ++c) {
      const float raw = reg[((size_t)a * hw + p) * 6 + c];
      o[c] = (raw - mu[c]) / (sd[c] + 1e-6f) * TGT[c];
    }
    const float wa = AW[a], ha = AH[a];
    const float cx = (float)ow + 0.5f, cy = (float)oh + 0.5f;
    const float xx = o[0] * wa + cx;
    const float yy = o[1] * ha + cy;
    const float ww = wa * expf(o[2]);
    const float hh = ha * expf(o[3]);
    const float dA = o[4] * ww;
    const float dB = o[5] * hh;

    const float v1x = xx + dA,        v1y = yy - hh * 0.5f;
    const float v2x = xx + ww * 0.5f, v2y = yy + dB;
    const float v3x = xx - dA,        v3y = yy + hh * 0.5f;
    const float v4x = xx - ww * 0.5f, v4y = yy - dB;

    float* vp = verts_ws + (size_t)k * 8;
    vp[0] = v1x; vp[1] = v1y; vp[2] = v2x; vp[3] = v2y;
    vp[4] = v3x; vp[5] = v3y; vp[6] = v4x; vp[7] = v4y;

    float xmin = fminf(fminf(v1x, v2x), fminf(v3x, v4x));
    float ymin = fminf(fminf(v1y, v2y), fminf(v3y, v4y));
    float xmax = fmaxf(fmaxf(v1x, v2x), fmaxf(v3x, v4x));
    float ymax = fmaxf(fmaxf(v1y, v2y), fmaxf(v3y, v4y));
    sHbb[k][0] = xmin; sHbb[k][1] = ymin; sHbb[k][2] = xmax; sHbb[k][3] = ymax;
    sKeep[k] = 1;
  }
  __syncthreads();

  // Phase B: greedy NMS (sequential i, parallel j > i)
  for (int i = 0; i < K; ++i) {
    if (sKeep[i]) {
      const float x1 = sHbb[i][0], y1 = sHbb[i][1], x2 = sHbb[i][2], y2 = sHbb[i][3];
      const float ai = fmaxf(x2 - x1, 0.0f) * fmaxf(y2 - y1, 0.0f);
      for (int j = i + 1 + tid; j < K; j += 256) {
        if (sKeep[j]) {
          const float bx1 = sHbb[j][0], by1 = sHbb[j][1];
          const float bx2 = sHbb[j][2], by2 = sHbb[j][3];
          const float aj = fmaxf(bx2 - bx1, 0.0f) * fmaxf(by2 - by1, 0.0f);
          const float iw = fmaxf(fminf(x2, bx2) - fmaxf(x1, bx1), 0.0f);
          const float ih = fmaxf(fminf(y2, by2) - fmaxf(y1, by1), 0.0f);
          const float inter = iw * ih;
          const float uni = fmaxf(ai + aj - inter, 1e-9f);
          if (inter / uni > 0.5f) sKeep[j] = 0;
        }
      }
    }
    __syncthreads();
  }

  // Phase C: write output (b, 6960, 9)
  for (int k = tid; k < K; k += 256) {
    const float kf = sKeep[k] ? 1.0f : 0.0f;
    const float* vp = verts_ws + (size_t)k * 8;
    float* op = out + ((size_t)bb * 6960 + out_off + k) * 9;
    #pragma unroll
    for (int c = 0; c < 8; ++c) op[c] = vp[c] * kf;
    op[8] = kscore[k] * kf;
  }
}

// ---------------------------------------------------------------------------
// Host launcher
// ---------------------------------------------------------------------------
extern "C" void kernel_launch(void* const* d_in, const int* in_sizes, int n_in,
                              void* d_out, int out_size, void* d_ws, size_t ws_size,
                              hipStream_t stream) {
  (void)in_sizes; (void)n_in; (void)out_size; (void)ws_size;

  const float* feats[5];
  for (int i = 0; i < 5; ++i) feats[i] = (const float*)d_in[i];
  const float* conv_w = (const float*)d_in[5];
  const float* conv_b = (const float*)d_in[6];
  const float* reg_w  = (const float*)d_in[7];
  const float* reg_b  = (const float*)d_in[8];
  const float* obj_w  = (const float*)d_in[9];
  const float* obj_b  = (const float*)d_in[10];
  float* out = (float*)d_out;
  float* ws  = (float*)d_ws;

  static const int HS[5]  = {128, 64, 32, 16, 8};
  static const int HW[5]  = {16384, 4096, 1024, 256, 64};
  static const int KS[5]  = {2000, 2000, 2000, 768, 192};
  static const int OFF[5] = {0, 2000, 4000, 6000, 6768};

  // Workspace layout (float slots), buffers reused across levels:
  const size_t X_OFF   = 0;                    // 512*16384         = 8388608
  const size_t REG_OFF = 8388608;              // 36*16384          =  589824
  const size_t OBJ_OFF = REG_OFF + 589824;     // 6*16384           =   98304
  const size_t ST_OFF  = OBJ_OFF + 98304;      // 2*6*2             =      24
  const size_t IDX_OFF = ST_OFF + 24;          // 2*2000 (ints)     =    4000
  const size_t SC_OFF  = IDX_OFF + 4000;       // 2*2000            =    4000
  const size_t VT_OFF  = SC_OFF + 4000;        // 2*2000*8          =   32000
  const size_t PW_OFF  = VT_OFF + 32000;       // 72*2*256*32 bf16  =  589824 floats

  float* X   = ws + X_OFF;
  float* REG = ws + REG_OFF;
  float* OBJ = ws + OBJ_OFF;
  float* ST  = ws + ST_OFF;
  int*   IDX = (int*)ws + IDX_OFF;
  float* SC  = ws + SC_OFF;
  float* VT  = ws + VT_OFF;
  void*  PW  = (void*)(ws + PW_OFF);

  for (int l = 0; l < 5; ++l) {
    const int H = HS[l], W = HS[l], hw = HW[l], K = KS[l];
    const int M = 2 * hw;

    prep_weights<<<dim3(72), 256, 0, stream>>>(
        conv_w + (size_t)l * 256 * 256 * 9, PW);

    conv3x3_wmma<<<dim3((M / 64) * 4), 256, 0, stream>>>(
        feats[l], PW, conv_b + (size_t)l * 256, X, H, W);

    heads_wmma<<<dim3(M / 64), 128, 0, stream>>>(
        X, reg_w + (size_t)l * 18 * 256, reg_b + (size_t)l * 18,
        obj_w + (size_t)l * 3 * 256, obj_b + (size_t)l * 3, REG, OBJ, hw);

    stats_kernel<<<dim3(12), 256, 0, stream>>>(REG, ST, hw);

    topk_kernel<<<dim3(2), 1024, 0, stream>>>(OBJ, IDX, SC, 3 * hw, K);

    decode_nms_kernel<<<dim3(2), 256, 0, stream>>>(
        REG, ST, IDX, SC, VT, out, hw, W, K, OFF[l]);
  }
}
